// DecoderBlock_55516747268180
// MI455X (gfx1250) — matrix operand order repaired and checked by reading, not compiled, not run on hardware
//
#include <hip/hip_runtime.h>
#include <hip/hip_bf16.h>
#include <math.h>

// ---------------------------------------------------------------------------
// Decoder block for MI455X (gfx1250, wave32, WMMA 16x16x32 bf16).
// All GEMMs run on v_wmma_f32_16x16x32_bf16 with fp32 accumulation.
// Weights are cast fp32->bf16 *pre-transposed*; GEMM uses double-buffered LDS
// fed by async global->LDS copies (ASYNCcnt). Attention peels the causal-mask
// work onto the diagonal chunk only.
// ---------------------------------------------------------------------------

#define HIDDEN 1024
#define NHEADS 16
#define HEADD  64
#define DFF    4096
#define BATCH  4
#define SEQ    2048
#define ROWS   (BATCH * SEQ)   // 8192

typedef __bf16 v16bf __attribute__((ext_vector_type(16)));
typedef float  v8f   __attribute__((ext_vector_type(8)));

union Frag16 { v16bf v; uint4 q[2]; };

// ---- async global->LDS (CDNA5) with sync fallback -------------------------
#if defined(__has_builtin)
#  if __has_builtin(__builtin_amdgcn_global_load_async_to_lds_b128)
#    define HAVE_ASYNC_LDS 1
#  endif
#endif
#ifndef HAVE_ASYNC_LDS
#  define HAVE_ASYNC_LDS 0
#endif

#if HAVE_ASYNC_LDS
typedef int v4i_a __attribute__((vector_size(16)));
typedef __attribute__((address_space(1))) v4i_a as1_v4i;
typedef __attribute__((address_space(3))) v4i_a as3_v4i;
#define ASYNC_CP16(dst, src)                                      \
  __builtin_amdgcn_global_load_async_to_lds_b128(                 \
      (as1_v4i*)(src), (as3_v4i*)(dst), 0, 0)
#endif

// ---------------------------------------------------------------------------
// Weight casts fp32 -> bf16, transposed: src [K][N] f32 -> dst [N][K] bf16
// ---------------------------------------------------------------------------
__global__ __launch_bounds__(256)
void cast_T_kernel(const float* __restrict__ src, __hip_bfloat16* __restrict__ dst,
                   int K, int N) {
  int idx = blockIdx.x * 256 + threadIdx.x;
  int k = idx / N, n = idx % N;
  dst[(size_t)n * K + k] = __float2bfloat16(src[idx]);
}

// Wq/Wk/Wv [H, C, D] -> WqkvT bf16 [3*C][C]; row = m*1024 + h*64 + d, col = c.
// Wq is pre-scaled by 1/sqrt(HEADD) = 0.125 (exact power-of-two in bf16).
__global__ __launch_bounds__(256)
void cast_qkv_kernel(const float* __restrict__ Wq, const float* __restrict__ Wk,
                     const float* __restrict__ Wv, __hip_bfloat16* __restrict__ WqkvT) {
  int idx = blockIdx.x * 256 + threadIdx.x;
  const int per = NHEADS * HIDDEN * HEADD;
  int m = idx / per;
  int rem = idx % per;
  int h = rem / (HIDDEN * HEADD);
  int c = (rem / HEADD) % HIDDEN;
  int d = rem % HEADD;
  const float* src = (m == 0) ? Wq : (m == 1) ? Wk : Wv;
  float val = src[rem];
  if (m == 0) val *= 0.125f;
  WqkvT[(size_t)(m * HIDDEN + h * HEADD + d) * HIDDEN + c] = __float2bfloat16(val);
}

// ---------------------------------------------------------------------------
// LayerNorm: one block (256 thr) per row of 1024, float4 loads, packed stores
// ---------------------------------------------------------------------------
__global__ __launch_bounds__(256)
void layernorm_kernel(const float* __restrict__ x, const float* __restrict__ g,
                      const float* __restrict__ be, __hip_bfloat16* __restrict__ out) {
  const int row = blockIdx.x;
  const float4 vv = ((const float4*)(x + (size_t)row * HIDDEN))[threadIdx.x];
  float s = vv.x + vv.y + vv.z + vv.w;
  float ss = vv.x * vv.x + vv.y * vv.y + vv.z * vv.z + vv.w * vv.w;
  __shared__ float red[2][8];
#pragma unroll
  for (int m = 1; m < 32; m <<= 1) {
    s += __shfl_xor(s, m, 32);
    ss += __shfl_xor(ss, m, 32);
  }
  const int wave = threadIdx.x >> 5, lane = threadIdx.x & 31;
  if (lane == 0) { red[0][wave] = s; red[1][wave] = ss; }
  __syncthreads();
  if (wave == 0) {
    s = red[0][lane & 7]; ss = red[1][lane & 7];
#pragma unroll
    for (int m = 1; m < 8; m <<= 1) {
      s += __shfl_xor(s, m, 32);
      ss += __shfl_xor(ss, m, 32);
    }
    if (lane == 0) { red[0][0] = s; red[1][0] = ss; }
  }
  __syncthreads();
  s = red[0][0]; ss = red[1][0];
  const float mu = s * (1.f / HIDDEN);
  const float var = ss * (1.f / HIDDEN) - mu * mu;
  const float inv = rsqrtf(var + 1e-5f);
  const float4 gg = ((const float4*)g)[threadIdx.x];
  const float4 bb = ((const float4*)be)[threadIdx.x];
  union { __hip_bfloat16 h[4]; uint2 u; } o;
  o.h[0] = __float2bfloat16((vv.x - mu) * inv * gg.x + bb.x);
  o.h[1] = __float2bfloat16((vv.y - mu) * inv * gg.y + bb.y);
  o.h[2] = __float2bfloat16((vv.z - mu) * inv * gg.z + bb.z);
  o.h[3] = __float2bfloat16((vv.w - mu) * inv * gg.w + bb.w);
  *(uint2*)&out[(size_t)row * HIDDEN + threadIdx.x * 4] = o.u;
}

// ---------------------------------------------------------------------------
// bf16 WMMA GEMM: C[M,N] = A[M,K] * B[K,N], B supplied pre-transposed BT[N][K].
// Block tile 128x128, 8 waves (4m x 2n), per-wave 32x64, K-chunk 64.
// Double-buffered LDS (rows padded to 72 elems -> conflict-free b128 gathers),
// staged by async global->LDS copies overlapped with the 16 WMMAs per chunk.
// Epilogue: +bias / relu / +fp32 residual / bf16-or-f32 out / optional C^T.
// ---------------------------------------------------------------------------
template <bool BIAS, bool RELU, bool RESID, bool OUTBF16, bool OUTTRANS>
__global__ __launch_bounds__(256)
void gemm_kernel(const __hip_bfloat16* __restrict__ A, int lda,
                 const __hip_bfloat16* __restrict__ BT, int ldb,
                 const float* __restrict__ bias,
                 const float* __restrict__ resid, int ldr,
                 void* __restrict__ Cout, int ldc,
                 int M, int N, int Klen) {
  __shared__ __hip_bfloat16 Al[2][128][72];
  __shared__ __hip_bfloat16 Bl[2][128][72];
  const int tid = threadIdx.x;
  const int wave = tid >> 5, lane = tid & 31;
  const int r = lane & 15, half = lane >> 4;
  const int wm = wave >> 1, wn = wave & 1;
  const int bm = blockIdx.y * 128, bn = blockIdx.x * 128;

  v8f acc[2][4] = {};

  const int srow = tid >> 1, scol = (tid & 1) * 32;   // 128 rows x 64 elems
  const __hip_bfloat16* arow_p = A + (size_t)(bm + srow) * lda + scol;
  const __hip_bfloat16* brow_p = BT + (size_t)(bn + srow) * ldb + scol;

  auto stage = [&](int buf, int kc) {
    const __hip_bfloat16* asrc = arow_p + kc;
    const __hip_bfloat16* bsrc = brow_p + kc;
#if HAVE_ASYNC_LDS
#pragma unroll
    for (int u = 0; u < 4; ++u)
      ASYNC_CP16(&Al[buf][srow][scol + u * 8], asrc + u * 8);
#pragma unroll
    for (int u = 0; u < 4; ++u)
      ASYNC_CP16(&Bl[buf][srow][scol + u * 8], bsrc + u * 8);
#else
#pragma unroll
    for (int u = 0; u < 4; ++u)
      *(uint4*)&Al[buf][srow][scol + u * 8] = *(const uint4*)(asrc + u * 8);
#pragma unroll
    for (int u = 0; u < 4; ++u)
      *(uint4*)&Bl[buf][srow][scol + u * 8] = *(const uint4*)(bsrc + u * 8);
#endif
  };

  stage(0, 0);
  int buf = 0;
  for (int kc = 0; kc < Klen; kc += 64) {
    if (kc + 64 < Klen) {
      stage(buf ^ 1, kc + 64);   // next tile in flight during compute
#if HAVE_ASYNC_LDS
      asm volatile("s_wait_asynccnt 0x8" ::: "memory");  // current tile resident
#endif
    } else {
#if HAVE_ASYNC_LDS
      asm volatile("s_wait_asynccnt 0x0" ::: "memory");
#endif
    }
    __syncthreads();

#pragma unroll
    for (int kk = 0; kk < 2; ++kk) {
      Frag16 af[2], bf[4];
#pragma unroll
      for (int mt = 0; mt < 2; ++mt) {
        const __hip_bfloat16* p = &Al[buf][wm * 32 + mt * 16 + r][kk * 32 + 8 * half];
        af[mt].q[0] = *(const uint4*)p;          // K = 8h .. 8h+7
        af[mt].q[1] = *(const uint4*)(p + 16);   // K = 16+8h .. 16+8h+7
      }
#pragma unroll
      for (int nt = 0; nt < 4; ++nt) {
        const __hip_bfloat16* p = &Bl[buf][wn * 64 + nt * 16 + r][kk * 32 + 8 * half];
        bf[nt].q[0] = *(const uint4*)p;          // K = 8h .. 8h+7
        bf[nt].q[1] = *(const uint4*)(p + 16);   // K = 16+8h .. 16+8h+7
      }
#pragma unroll
      for (int mt = 0; mt < 2; ++mt)
#pragma unroll
        for (int nt = 0; nt < 4; ++nt)
          acc[mt][nt] = __builtin_amdgcn_wmma_f32_16x16x32_bf16(
              false, af[mt].v, false, bf[nt].v, (short)0, acc[mt][nt], false, false);
    }
    __syncthreads();   // all reads done before this buffer is restaged
    buf ^= 1;
  }

  // --- epilogue ---
#pragma unroll
  for (int mt = 0; mt < 2; ++mt)
#pragma unroll
    for (int nt = 0; nt < 4; ++nt)
#pragma unroll
      for (int j = 0; j < 8; ++j) {
        int row = bm + wm * 32 + mt * 16 + j + 8 * half;
        int col = bn + wn * 64 + nt * 16 + r;
        float v = acc[mt][nt][j];
        if (BIAS) v += bias[col];
        if (RELU) v = fmaxf(v, 0.f);
        if (RESID) v += resid[(size_t)row * ldr + col];
        size_t idx = OUTTRANS ? (size_t)col * ldc + row : (size_t)row * ldc + col;
        if (OUTBF16)
          ((__hip_bfloat16*)Cout)[idx] = __float2bfloat16(v);
        else
          ((float*)Cout)[idx] = v;
      }
}

// ---------------------------------------------------------------------------
// Causal flash attention.
//   qk : bf16 [ROWS, 2048]  (Q | K, per-head 64-col slabs; Q pre-scaled)
//   vT : bf16 [1024, ROWS]  (row = h*64+d, col = b*T+t)
// Block = 128 thr (4 waves); each wave owns a 16-row Q tile, streams K/V in
// 32-chunks with online softmax. Mask/compare work only on the diagonal chunk.
// ---------------------------------------------------------------------------
template <bool DIAG>
__device__ __forceinline__ void attn_chunk(
    int s0, int m0, int r, int half,
    const __hip_bfloat16* __restrict__ Kp, int ldq,
    const __hip_bfloat16* __restrict__ vTh,       // + (d)*ROWS + t indexing
    const Frag16* qf, v8f* of, float* rm, float* rl,
    __hip_bfloat16 (*PtW)[40]) {
  // S = Q * K^T  (K rows loaded directly as transposed-B fragments)
  v8f sf[2] = {};
#pragma unroll
  for (int nt = 0; nt < 2; ++nt)
#pragma unroll
    for (int c = 0; c < 2; ++c) {
      Frag16 kf;
      const __hip_bfloat16* p =
          Kp + (size_t)(s0 + nt * 16 + r) * ldq + c * 32 + 8 * half;
      kf.q[0] = *(const uint4*)p;
      kf.q[1] = *(const uint4*)(p + 16);
      sf[nt] = __builtin_amdgcn_wmma_f32_16x16x32_bf16(
          false, qf[c].v, false, kf.v, (short)0, sf[nt], false, false);
    }

  // causal mask only on the diagonal chunk; chunk row-max
  float cm[8];
#pragma unroll
  for (int j = 0; j < 8; ++j) cm[j] = -INFINITY;
#pragma unroll
  for (int nt = 0; nt < 2; ++nt)
#pragma unroll
    for (int j = 0; j < 8; ++j) {
      float s = sf[nt][j];
      if (DIAG) {
        int col = s0 + nt * 16 + r;
        int row = m0 + j + 8 * half;
        if (col > row) s = -INFINITY;
        sf[nt][j] = s;
      }
      cm[j] = fmaxf(cm[j], s);
    }
#pragma unroll
  for (int j = 0; j < 8; ++j)
#pragma unroll
    for (int m = 1; m < 16; m <<= 1)
      cm[j] = fmaxf(cm[j], __shfl_xor(cm[j], m, 32));

  // online-softmax rescale
  float csum[8];
#pragma unroll
  for (int j = 0; j < 8; ++j) {
    float mn = fmaxf(rm[j], cm[j]);
    float sc = __expf(rm[j] - mn);
    rm[j] = mn;
    rl[j] *= sc;
#pragma unroll
    for (int dt = 0; dt < 4; ++dt) of[dt][j] *= sc;
    csum[j] = 0.f;
  }

  // P = exp(S - m), stage to LDS for lane transpose
#pragma unroll
  for (int nt = 0; nt < 2; ++nt)
#pragma unroll
    for (int j = 0; j < 8; ++j) {
      float p = __expf(sf[nt][j] - rm[j]);
      csum[j] += p;
      PtW[j + 8 * half][nt * 16 + r] = __float2bfloat16(p);
    }
#pragma unroll
  for (int j = 0; j < 8; ++j) {
#pragma unroll
    for (int m = 1; m < 16; m <<= 1) csum[j] += __shfl_xor(csum[j], m, 32);
    rl[j] += csum[j];
  }

  // intra-wave LDS ordering (same-wave DS ops retire in order)
  asm volatile("s_wait_dscnt 0" ::: "memory");

  // P as A fragment
  Frag16 pf;
  {
    const __hip_bfloat16* p = &PtW[r][8 * half];
    pf.q[0] = *(const uint4*)p;
    pf.q[1] = *(const uint4*)(p + 16);
  }
  // O += P * V : V^T fragments straight from global (b128 x2 per tile)
#pragma unroll
  for (int dt = 0; dt < 4; ++dt) {
    Frag16 vf;
    const __hip_bfloat16* p = vTh + (size_t)(dt * 16 + r) * ROWS + s0 + 8 * half;
    vf.q[0] = *(const uint4*)p;
    vf.q[1] = *(const uint4*)(p + 16);
    of[dt] = __builtin_amdgcn_wmma_f32_16x16x32_bf16(
        false, pf.v, false, vf.v, (short)0, of[dt], false, false);
  }
}

__global__ __launch_bounds__(128)
void attn_kernel(const __hip_bfloat16* __restrict__ qk,
                 const __hip_bfloat16* __restrict__ vT,
                 __hip_bfloat16* __restrict__ aout) {
  const int bh = blockIdx.y;
  const int b = bh / NHEADS, h = bh % NHEADS;
  const int wave = threadIdx.x >> 5, lane = threadIdx.x & 31;
  const int r = lane & 15, half = lane >> 4;
  const int m0 = blockIdx.x * 64 + wave * 16;     // q-row tile base within SEQ
  const int ldq = 2 * HIDDEN;
  const size_t rb = (size_t)b * SEQ;
  const __hip_bfloat16* Qp = qk + rb * ldq + h * HEADD;
  const __hip_bfloat16* Kp = qk + rb * ldq + HIDDEN + h * HEADD;
  const __hip_bfloat16* vTh = vT + (size_t)(h * HEADD) * ROWS + rb;

  __shared__ __hip_bfloat16 Pt[4][16][40];   // per-wave P tile [m][k]

  // Q fragments (A layout), rows m0..m0+15, k-chunks 0..31 / 32..63
  Frag16 qf[2];
#pragma unroll
  for (int c = 0; c < 2; ++c) {
    const __hip_bfloat16* p = Qp + (size_t)(m0 + r) * ldq + c * 32 + 8 * half;
    qf[c].q[0] = *(const uint4*)p;
    qf[c].q[1] = *(const uint4*)(p + 16);
  }

  v8f of[4] = {};
  float rm[8], rl[8];
#pragma unroll
  for (int j = 0; j < 8; ++j) { rm[j] = -INFINITY; rl[j] = 0.f; }

  const int diag0 = (m0 >> 5) * 32;
  for (int s0 = 0; s0 < diag0; s0 += 32)
    attn_chunk<false>(s0, m0, r, half, Kp, ldq, vTh, qf, of, rm, rl, Pt[wave]);
  attn_chunk<true>(diag0, m0, r, half, Kp, ldq, vTh, qf, of, rm, rl, Pt[wave]);

  // normalize and write concat-head output (bf16, cols h*64 + d)
#pragma unroll
  for (int dt = 0; dt < 4; ++dt)
#pragma unroll
    for (int j = 0; j < 8; ++j) {
      float v = of[dt][j] / rl[j];
      size_t row = rb + m0 + j + 8 * half;
      aout[row * HIDDEN + h * HEADD + dt * 16 + r] = __float2bfloat16(v);
    }
}

// ---------------------------------------------------------------------------
// Launcher
// ---------------------------------------------------------------------------
extern "C" void kernel_launch(void* const* d_in, const int* in_sizes, int n_in,
                              void* d_out, int out_size, void* d_ws, size_t ws_size,
                              hipStream_t stream) {
  const float* x   = (const float*)d_in[0];
  const float* Wq  = (const float*)d_in[1];
  const float* Wk  = (const float*)d_in[2];
  const float* Wv  = (const float*)d_in[3];
  const float* Wo  = (const float*)d_in[4];
  const float* bo  = (const float*)d_in[5];
  const float* W1  = (const float*)d_in[6];
  const float* b1  = (const float*)d_in[7];
  const float* W2  = (const float*)d_in[8];
  const float* b2  = (const float*)d_in[9];
  const float* g1  = (const float*)d_in[10];
  const float* be1 = (const float*)d_in[11];
  const float* g2  = (const float*)d_in[12];
  const float* be2 = (const float*)d_in[13];
  float* out = (float*)d_out;

  // workspace layout (136 MB; ff1 aliases qk+vT+aout, h2 aliases h1)
  char* w = (char*)d_ws;
  size_t off = 0;
  auto take = [&](size_t n) { void* p = w + off; off += (n + 255) & ~(size_t)255; return p; };
  __hip_bfloat16* WqkvT_b = (__hip_bfloat16*)take((size_t)3 * HIDDEN * HIDDEN * 2); // 6 MB  [3072][1024]
  __hip_bfloat16* WoT_b   = (__hip_bfloat16*)take((size_t)HIDDEN * HIDDEN * 2);     // 2 MB  [1024][1024]
  __hip_bfloat16* W1T_b   = (__hip_bfloat16*)take((size_t)DFF * HIDDEN * 2);        // 8 MB  [4096][1024]
  __hip_bfloat16* W2T_b   = (__hip_bfloat16*)take((size_t)HIDDEN * DFF * 2);        // 8 MB  [1024][4096]
  __hip_bfloat16* h1_b    = (__hip_bfloat16*)take((size_t)ROWS * 2 * HIDDEN);       // 16 MB
  __hip_bfloat16* qk_b    = (__hip_bfloat16*)take((size_t)ROWS * 2 * HIDDEN * 2);   // 32 MB [8192][2048]
  __hip_bfloat16* vT_b    = (__hip_bfloat16*)take((size_t)HIDDEN * ROWS * 2);       // 16 MB [1024][8192]
  __hip_bfloat16* aout_b  = (__hip_bfloat16*)take((size_t)ROWS * HIDDEN * 2);       // 16 MB
  float*          x2      = (float*)take((size_t)ROWS * HIDDEN * 4);                // 32 MB
  __hip_bfloat16* h2_b  = h1_b;   // h1 dead after QK/V GEMMs
  __hip_bfloat16* ff1_b = qk_b;   // qk+vT+aout (64 MB) dead after Wo GEMM

  // 1) weight casts (transposed; Wq pre-scaled by 0.125)
  cast_qkv_kernel<<<(3 * NHEADS * HIDDEN * HEADD) / 256, 256, 0, stream>>>(Wq, Wk, Wv, WqkvT_b);
  cast_T_kernel<<<(HIDDEN * HIDDEN) / 256, 256, 0, stream>>>(Wo, WoT_b, HIDDEN, HIDDEN);
  cast_T_kernel<<<(HIDDEN * DFF) / 256, 256, 0, stream>>>(W1, W1T_b, HIDDEN, DFF);
  cast_T_kernel<<<(DFF * HIDDEN) / 256, 256, 0, stream>>>(W2, W2T_b, DFF, HIDDEN);

  // 2) h1 = LN1(x)
  layernorm_kernel<<<ROWS, 256, 0, stream>>>(x, g1, be1, h1_b);

  // 3a) qk = h1 @ Wqk   [8192 x 2048]
  gemm_kernel<false, false, false, true, false>
      <<<dim3(2 * HIDDEN / 128, ROWS / 128), 256, 0, stream>>>(
          h1_b, HIDDEN, WqkvT_b, HIDDEN, nullptr, nullptr, 0,
          qk_b, 2 * HIDDEN, ROWS, 2 * HIDDEN, HIDDEN);
  // 3b) vT = (h1 @ Wv)^T   [1024 x 8192]
  gemm_kernel<false, false, false, true, true>
      <<<dim3(HIDDEN / 128, ROWS / 128), 256, 0, stream>>>(
          h1_b, HIDDEN, WqkvT_b + (size_t)2 * HIDDEN * HIDDEN, HIDDEN,
          nullptr, nullptr, 0, vT_b, ROWS, ROWS, HIDDEN, HIDDEN);

  // 4) causal attention -> aout (bf16, heads concatenated)
  attn_kernel<<<dim3(SEQ / 64, BATCH * NHEADS), 128, 0, stream>>>(qk_b, vT_b, aout_b);

  // 5) x2 = aout @ Wo + bo + x
  gemm_kernel<true, false, true, false, false>
      <<<dim3(HIDDEN / 128, ROWS / 128), 256, 0, stream>>>(
          aout_b, HIDDEN, WoT_b, HIDDEN, bo, x, HIDDEN,
          x2, HIDDEN, ROWS, HIDDEN, HIDDEN);

  // 6) h2 = LN2(x2)
  layernorm_kernel<<<ROWS, 256, 0, stream>>>(x2, g2, be2, h2_b);

  // 7) ff1 = relu(h2 @ W1 + b1)   [8192 x 4096]
  gemm_kernel<true, true, false, true, false>
      <<<dim3(DFF / 128, ROWS / 128), 256, 0, stream>>>(
          h2_b, HIDDEN, W1T_b, HIDDEN, b1, nullptr, 0,
          ff1_b, DFF, ROWS, DFF, HIDDEN);

  // 8) out = ff1 @ W2 + b2 + x2
  gemm_kernel<true, false, true, false, false>
      <<<dim3(HIDDEN / 128, ROWS / 128), 256, 0, stream>>>(
          ff1_b, DFF, W2T_b, DFF, b2, x2, HIDDEN,
          out, HIDDEN, ROWS, HIDDEN, DFF);
}
